// PointNet2Stage_37812892074202
// MI455X (gfx1250) — compile-verified, matrix operand order checked by reading, not run on hardware
//
#include <hip/hip_runtime.h>
#include <hip/hip_bf16.h>
#include <stdint.h>

typedef __bf16 bf16_t;
typedef __attribute__((ext_vector_type(8)))  __bf16 bf16x8;
typedef __attribute__((ext_vector_type(16))) __bf16 bf16x16;
typedef __attribute__((ext_vector_type(8)))  float  f32x8;
typedef __attribute__((ext_vector_type(4)))  unsigned int u32x4;
typedef __attribute__((ext_vector_type(8)))  int i32x8;
typedef __attribute__((ext_vector_type(4)))  int i32x4;

#define NPTS 524288
#define NVOX 32768

#define DEV __device__ __forceinline__

DEV bf16x16 frag_from_chunks(bf16x8 c0, bf16x8 c1) {
    bf16x16 r;
#pragma unroll
    for (int i = 0; i < 8; ++i) { r[i] = c0[i]; r[8 + i] = c1[i]; }
    return r;
}

DEV bf16x8 load8_f32_as_bf16(const float* p) {
    bf16x8 r;
#pragma unroll
    for (int i = 0; i < 8; ++i) r[i] = (__bf16)p[i];
    return r;
}

DEV float relu(float v) { return v > 0.f ? v : 0.f; }

// ---------------------------------------------------------------------------
// Tensor Data Mover: DMA a contiguous [rows][cols] bf16 block (pre-transposed
// weights) from global into LDS. 2D tensor, tile == tensor. Issued by one
// wave; EXEC is ignored by TDM. Completion tracked with TENSORcnt.
// This toolchain exposes the 6-arg builtin: (g0, g1, g2lo, g2hi, g3, cpol).
// ---------------------------------------------------------------------------
DEV void tdm_load_weights(const bf16_t* gsrc, bf16_t* lds_dst, int cols, int rows) {
    uint64_t ga = (uint64_t)(uintptr_t)gsrc;
    // generic LDS pointer: aperture lives in bits [63:32]; low 32 = LDS offset
    uint32_t lds_off = (uint32_t)(uintptr_t)lds_dst;
    u32x4 g0;
    g0[0] = 1u;                                   // count=1 valid descriptor
    g0[1] = lds_off;                              // lds_addr (bytes)
    g0[2] = (uint32_t)ga;                         // global_addr[31:0]
    g0[3] = ((uint32_t)(ga >> 32) & 0x01FFFFFFu)  // global_addr[56:32]
            | (2u << 30);                         // type = 2 ("image")
    i32x8 g1;
    g1[0] = (int)(1u << 16);                      // wg_mask=0, data_size=1 (2B)
    g1[1] = (int)((cols & 0xFFFF) << 16);         // tensor_dim0[15:0]
    g1[2] = (int)(((cols >> 16) & 0xFFFF) | ((rows & 0xFFFF) << 16)); // dim0 hi | dim1 lo
    g1[3] = (int)(((rows >> 16) & 0xFFFF) | ((cols & 0xFFFF) << 16)); // dim1 hi | tile_dim0
    g1[4] = (int)(rows & 0xFFFF);                 // tile_dim1 (tile_dim2 = 0)
    g1[5] = cols;                                 // tensor_dim0_stride[31:0]
    g1[6] = 0;
    g1[7] = 0;
    i32x4 z4 = {0, 0, 0, 0};
    i32x8 z8 = {0, 0, 0, 0, 0, 0, 0, 0};
    __builtin_amdgcn_tensor_load_to_lds(g0, g1, z4, z4, z8, 0);
}

DEV void tdm_wait() { __builtin_amdgcn_s_wait_tensorcnt(0); }

// ---------------------------------------------------------------------------
// Zero-init the two scatter-max accumulators (identity 0 == relu lower bound)
// ---------------------------------------------------------------------------
__global__ void __launch_bounds__(256) k_zero(uint32_t* __restrict__ p, int n) {
    int i = blockIdx.x * 256 + threadIdx.x;
    int stride = gridDim.x * 256;
    for (; i < n; i += stride) p[i] = 0u;
}

// ---------------------------------------------------------------------------
// One-shot weight prep: dst[n*K+k] = bf16(src[k*N+n])   (transpose + convert)
// ---------------------------------------------------------------------------
__global__ void __launch_bounds__(256)
k_transpose_bf16(const float* __restrict__ src, bf16_t* __restrict__ dst, int K, int N) {
    int i = blockIdx.x * 256 + threadIdx.x;
    if (i >= K * N) return;
    int n = i / K, k = i - n * K;
    dst[i] = (__bf16)src[k * N + n];
}

// ---------------------------------------------------------------------------
// Stage 1+2: pf1 = relu(x@W1+b1) [scalar], pf2 = relu(pf1@W2+b2) [WMMA],
// store pf2 bf16 and scatter-max (atomicMax on uint bits) into voxbuf1.
// W2^T (bf16, pre-transposed) arrives via TDM, overlapped with stage 1.
// ---------------------------------------------------------------------------
__global__ void __launch_bounds__(256)
k_point_mlp12(const float* __restrict__ inp, const long long* __restrict__ idx,
              const float* __restrict__ W1, const float* __restrict__ b1,
              const bf16_t* __restrict__ Wt2, const float* __restrict__ b2,
              bf16_t* __restrict__ pf2, unsigned int* __restrict__ voxbuf1) {
    __shared__ float  W1s[6 * 64];
    __shared__ float  b1s[64];
    __shared__ bf16_t W2t[128 * 64];   // filled by TDM
    __shared__ bf16_t pf1[256 * 64];
    const int tid = threadIdx.x;
    if (tid < 32) tdm_load_weights(Wt2, W2t, 64, 128);   // async 16KB DMA
    for (int i = tid; i < 6 * 64; i += 256) W1s[i] = W1[i];
    if (tid < 64) b1s[tid] = b1[tid];
    __syncthreads();

    // stage 1: one point per thread (overlaps with TDM)
    const int p = blockIdx.x * 256 + tid;
    float x[6];
#pragma unroll
    for (int c = 0; c < 6; ++c) x[c] = inp[p * 6 + c];
    for (int o = 0; o < 64; ++o) {
        float a = b1s[o];
#pragma unroll
        for (int c = 0; c < 6; ++c) a = fmaf(x[c], W1s[c * 64 + o], a);
        pf1[tid * 64 + o] = (__bf16)relu(a);
    }
    if (tid < 32) tdm_wait();
    __syncthreads();

    // stage 2: WMMA, each wave: two 16-row strips x 8 col tiles, K=64
    const int wave = tid >> 5, lane = tid & 31;
    const int m = lane & 15, g = lane >> 4;
    for (int s = 0; s < 2; ++s) {
        const int strip = wave * 2 + s;
        const int row0 = strip * 16;
        const bf16_t* arow = pf1 + (row0 + m) * 64;
        bf16x16 afrag[2];
#pragma unroll
        for (int ks = 0; ks < 2; ++ks) {
            int kb = ks * 32;
            afrag[ks] = frag_from_chunks(*(const bf16x8*)(arow + kb + g * 8),
                                         *(const bf16x8*)(arow + kb + 16 + g * 8));
        }
        int vr[8];
#pragma unroll
        for (int r = 0; r < 8; ++r)
            vr[r] = (int)idx[blockIdx.x * 256 + row0 + r + 8 * g];
#pragma unroll 1
        for (int j = 0; j < 8; ++j) {
            f32x8 acc = {};
            const bf16_t* wrow = W2t + (j * 16 + m) * 64;
#pragma unroll
            for (int ks = 0; ks < 2; ++ks) {
                bf16x16 bfrag = frag_from_chunks(*(const bf16x8*)(wrow + ks * 32 + g * 8),
                                                 *(const bf16x8*)(wrow + ks * 32 + 16 + g * 8));
                acc = __builtin_amdgcn_wmma_f32_16x16x32_bf16(false, afrag[ks], false, bfrag,
                                                              (short)0, acc, false, false);
            }
            const int n = j * 16 + m;
            const float bias = b2[n];
#pragma unroll
            for (int r = 0; r < 8; ++r) {
                float v = relu(acc[r] + bias);
                int rowp = blockIdx.x * 256 + row0 + r + 8 * g;
                pf2[rowp * 128 + n] = (__bf16)v;
                atomicMax(&voxbuf1[vr[r] * 128 + n], __float_as_uint(v));
            }
        }
    }
}

// ---------------------------------------------------------------------------
// voxfeat1 = relu(voxbuf1 @ Wv1 + bv1), [32768,128]@[128,128], bf16 out.
// ---------------------------------------------------------------------------
__global__ void __launch_bounds__(256)
k_vox_mlp1(const float* __restrict__ voxbuf1, const bf16_t* __restrict__ Wtv1,
           const float* __restrict__ bv1, bf16_t* __restrict__ voxfeat1) {
    __shared__ bf16_t Wt[128 * 128];
    const int tid = threadIdx.x;
    if (tid < 32) tdm_load_weights(Wtv1, Wt, 128, 128);   // async 32KB DMA
    const int wave = tid >> 5, lane = tid & 31, m = lane & 15, g = lane >> 4;
    const int row0 = blockIdx.x * 128 + wave * 16;
    const float* arow = voxbuf1 + (row0 + m) * 128;
    bf16x16 afrag[4];
#pragma unroll
    for (int ks = 0; ks < 4; ++ks) {
        int kb = ks * 32;
        afrag[ks] = frag_from_chunks(load8_f32_as_bf16(arow + kb + g * 8),
                                     load8_f32_as_bf16(arow + kb + 16 + g * 8));
    }
    if (tid < 32) tdm_wait();
    __syncthreads();
#pragma unroll 1
    for (int j = 0; j < 8; ++j) {
        f32x8 acc = {};
        const bf16_t* wrow = Wt + (j * 16 + m) * 128;
#pragma unroll
        for (int ks = 0; ks < 4; ++ks) {
            bf16x16 bfrag = frag_from_chunks(*(const bf16x8*)(wrow + ks * 32 + g * 8),
                                             *(const bf16x8*)(wrow + ks * 32 + 16 + g * 8));
            acc = __builtin_amdgcn_wmma_f32_16x16x32_bf16(false, afrag[ks], false, bfrag,
                                                          (short)0, acc, false, false);
        }
        int n = j * 16 + m;
        float bias = bv1[n];
#pragma unroll
        for (int r = 0; r < 8; ++r) {
            float v = relu(acc[r] + bias);
            voxfeat1[(row0 + r + 8 * g) * 128 + n] = (__bf16)v;
        }
    }
}

// ---------------------------------------------------------------------------
// pf4 = relu(concat(voxfeat1[idx], pf2) @ W3 + b3). A gathered on the fly.
// W3^T via TDM into 128KB dynamic LDS; C tiles staged in LDS for coalesced
// global_store_b128.
// ---------------------------------------------------------------------------
__global__ void __launch_bounds__(256)
k_point_mlp3(const bf16_t* __restrict__ pf2, const bf16_t* __restrict__ voxf1,
             const long long* __restrict__ idx,
             const bf16_t* __restrict__ Wt3, const float* __restrict__ b3,
             bf16_t* __restrict__ pf4) {
    extern __shared__ bf16_t Wt[];            // [256][256] bf16 + 8*256 staging
    const int tid = threadIdx.x;
    if (tid < 32) tdm_load_weights(Wt3, Wt, 256, 256);    // async 128KB DMA
    const int wave = tid >> 5, lane = tid & 31, m = lane & 15, g = lane >> 4;
    const int row0 = blockIdx.x * 128 + wave * 16;
    const int p = row0 + m;
    const bf16_t* vbase = voxf1 + (int)idx[p] * 128;
    const bf16_t* pbase = pf2 + p * 128;
    bf16x16 afrag[8];
#pragma unroll
    for (int ks = 0; ks < 8; ++ks) {
        int k0 = ks * 32 + g * 8;
        int k1 = ks * 32 + 16 + g * 8;
        const bf16_t* s0 = (k0 < 128) ? (vbase + k0) : (pbase + k0 - 128);
        const bf16_t* s1 = (k1 < 128) ? (vbase + k1) : (pbase + k1 - 128);
        afrag[ks] = frag_from_chunks(*(const bf16x8*)s0, *(const bf16x8*)s1);
    }
    if (tid < 32) tdm_wait();
    __syncthreads();
    bf16_t* stg = Wt + 256 * 256 + wave * 256;   // 16x16 bf16 per-wave staging
#pragma unroll 1
    for (int j = 0; j < 16; ++j) {
        f32x8 acc = {};
        const bf16_t* wrow = Wt + ((j * 16 + m) << 8);
#pragma unroll
        for (int ks = 0; ks < 8; ++ks) {
            bf16x16 bfrag = frag_from_chunks(*(const bf16x8*)(wrow + ks * 32 + g * 8),
                                             *(const bf16x8*)(wrow + ks * 32 + 16 + g * 8));
            acc = __builtin_amdgcn_wmma_f32_16x16x32_bf16(false, afrag[ks], false, bfrag,
                                                          (short)0, acc, false, false);
        }
        int n = j * 16 + m;
        float bias = b3[n];
#pragma unroll
        for (int r = 0; r < 8; ++r)
            stg[(r + 8 * g) * 16 + m] = (__bf16)relu(acc[r] + bias);
        asm volatile("s_wait_dscnt 0x0" ::: "memory");    // wave-local LDS RAW
        int rr = lane >> 1, seg = lane & 1;               // 2 lanes cover a row
        bf16x8 vrow = *(const bf16x8*)(stg + rr * 16 + seg * 8);
        *(bf16x8*)(pf4 + (row0 + rr) * 256 + j * 16 + seg * 8) = vrow;  // b128
    }
}

// ---------------------------------------------------------------------------
// pf5 = relu(pf4 @ W4 + b4), scatter-max into vox2buf (no pf5 in memory).
// ---------------------------------------------------------------------------
__global__ void __launch_bounds__(256)
k_point_mlp4(const bf16_t* __restrict__ pf4, const long long* __restrict__ idx,
             const bf16_t* __restrict__ Wt4, const float* __restrict__ b4,
             unsigned int* __restrict__ vox2) {
    extern __shared__ bf16_t Wt[];
    const int tid = threadIdx.x;
    if (tid < 32) tdm_load_weights(Wt4, Wt, 256, 256);
    const int wave = tid >> 5, lane = tid & 31, m = lane & 15, g = lane >> 4;
    const int row0 = blockIdx.x * 128 + wave * 16;
    const bf16_t* arow = pf4 + (row0 + m) * 256;
    bf16x16 afrag[8];
#pragma unroll
    for (int ks = 0; ks < 8; ++ks) {
        int kb = ks * 32;
        afrag[ks] = frag_from_chunks(*(const bf16x8*)(arow + kb + g * 8),
                                     *(const bf16x8*)(arow + kb + 16 + g * 8));
    }
    int vr[8];
#pragma unroll
    for (int r = 0; r < 8; ++r) vr[r] = (int)idx[row0 + r + 8 * g];
    if (tid < 32) tdm_wait();
    __syncthreads();
#pragma unroll 1
    for (int j = 0; j < 16; ++j) {
        f32x8 acc = {};
        const bf16_t* wrow = Wt + ((j * 16 + m) << 8);
#pragma unroll
        for (int ks = 0; ks < 8; ++ks) {
            bf16x16 bfrag = frag_from_chunks(*(const bf16x8*)(wrow + ks * 32 + g * 8),
                                             *(const bf16x8*)(wrow + ks * 32 + 16 + g * 8));
            acc = __builtin_amdgcn_wmma_f32_16x16x32_bf16(false, afrag[ks], false, bfrag,
                                                          (short)0, acc, false, false);
        }
        int n = j * 16 + m;
        float bias = b4[n];
#pragma unroll
        for (int r = 0; r < 8; ++r) {
            float v = relu(acc[r] + bias);
            atomicMax(&vox2[vr[r] * 256 + n], __float_as_uint(v));
        }
    }
}

// ---------------------------------------------------------------------------
// out = relu(vox2buf @ Wv2 + bv2), f32 output. [32768,256]@[256,256].
// ---------------------------------------------------------------------------
__global__ void __launch_bounds__(256)
k_vox_mlp2(const float* __restrict__ vox2, const bf16_t* __restrict__ Wtv2,
           const float* __restrict__ bv2, float* __restrict__ out) {
    extern __shared__ bf16_t Wt[];
    const int tid = threadIdx.x;
    if (tid < 32) tdm_load_weights(Wtv2, Wt, 256, 256);
    const int wave = tid >> 5, lane = tid & 31, m = lane & 15, g = lane >> 4;
    const int row0 = blockIdx.x * 128 + wave * 16;
    const float* arow = vox2 + (row0 + m) * 256;
    bf16x16 afrag[8];
#pragma unroll
    for (int ks = 0; ks < 8; ++ks) {
        int kb = ks * 32;
        afrag[ks] = frag_from_chunks(load8_f32_as_bf16(arow + kb + g * 8),
                                     load8_f32_as_bf16(arow + kb + 16 + g * 8));
    }
    if (tid < 32) tdm_wait();
    __syncthreads();
#pragma unroll 1
    for (int j = 0; j < 16; ++j) {
        f32x8 acc = {};
        const bf16_t* wrow = Wt + ((j * 16 + m) << 8);
#pragma unroll
        for (int ks = 0; ks < 8; ++ks) {
            bf16x16 bfrag = frag_from_chunks(*(const bf16x8*)(wrow + ks * 32 + g * 8),
                                             *(const bf16x8*)(wrow + ks * 32 + 16 + g * 8));
            acc = __builtin_amdgcn_wmma_f32_16x16x32_bf16(false, afrag[ks], false, bfrag,
                                                          (short)0, acc, false, false);
        }
        int n = j * 16 + m;
        float bias = bv2[n];
#pragma unroll
        for (int r = 0; r < 8; ++r)
            out[(row0 + r + 8 * g) * 256 + n] = relu(acc[r] + bias);
    }
}

// ---------------------------------------------------------------------------
extern "C" void kernel_launch(void* const* d_in, const int* in_sizes, int n_in,
                              void* d_out, int out_size, void* d_ws, size_t ws_size,
                              hipStream_t stream) {
    const float*     inp = (const float*)d_in[0];
    const long long* idx = (const long long*)d_in[1];   // int64 indices
    const float* W1  = (const float*)d_in[3];
    const float* b1  = (const float*)d_in[4];
    const float* W2  = (const float*)d_in[5];
    const float* b2  = (const float*)d_in[6];
    const float* Wv1 = (const float*)d_in[7];
    const float* bv1 = (const float*)d_in[8];
    const float* W3  = (const float*)d_in[9];
    const float* b3  = (const float*)d_in[10];
    const float* W4  = (const float*)d_in[11];
    const float* b4  = (const float*)d_in[12];
    const float* Wv2 = (const float*)d_in[13];
    const float* bv2 = (const float*)d_in[14];

    // workspace layout (all 16B-aligned)
    char* ws = (char*)d_ws;
    float*  voxbuf1  = (float*)(ws);                                        // [V,128] f32
    float*  vox2buf  = (float*)(ws + (size_t)NVOX * 128 * 4);               // [V,256] f32
    bf16_t* voxfeat1 = (bf16_t*)((char*)vox2buf + (size_t)NVOX * 256 * 4);  // [V,128] bf16
    bf16_t* pf2      = (bf16_t*)((char*)voxfeat1 + (size_t)NVOX * 128 * 2); // [N,128] bf16
    bf16_t* pf4      = (bf16_t*)((char*)pf2 + (size_t)NPTS * 128 * 2);      // [N,256] bf16
    bf16_t* Wt2      = (bf16_t*)((char*)pf4 + (size_t)NPTS * 256 * 2);      // [128][64]
    bf16_t* Wtv1     = Wt2 + 128 * 64;                                      // [128][128]
    bf16_t* Wt3      = Wtv1 + 128 * 128;                                    // [256][256]
    bf16_t* Wt4      = Wt3 + 256 * 256;                                     // [256][256]
    bf16_t* Wtv2     = Wt4 + 256 * 256;                                     // [256][256]
    float*  out      = (float*)d_out;                                       // [V,256] f32

    const int zero_n = NVOX * 128 + NVOX * 256;  // both accumulators contiguous
    k_zero<<<4096, 256, 0, stream>>>((uint32_t*)voxbuf1, zero_n);

    // one-shot weight transpose+convert into workspace (bf16)
    k_transpose_bf16<<<(128 * 64 + 255) / 256, 256, 0, stream>>>(W2, Wt2, 64, 128);
    k_transpose_bf16<<<(128 * 128 + 255) / 256, 256, 0, stream>>>(Wv1, Wtv1, 128, 128);
    k_transpose_bf16<<<(256 * 256 + 255) / 256, 256, 0, stream>>>(W3, Wt3, 256, 256);
    k_transpose_bf16<<<(256 * 256 + 255) / 256, 256, 0, stream>>>(W4, Wt4, 256, 256);
    k_transpose_bf16<<<(256 * 256 + 255) / 256, 256, 0, stream>>>(Wv2, Wtv2, 256, 256);

    k_point_mlp12<<<NPTS / 256, 256, 0, stream>>>(inp, idx, W1, b1, Wt2, b2,
                                                  pf2, (unsigned int*)voxbuf1);
    k_vox_mlp1<<<NVOX / 128, 256, 0, stream>>>(voxbuf1, Wtv1, bv1, voxfeat1);
    k_point_mlp3<<<NPTS / 128, 256, (256 * 256 + 8 * 256) * 2, stream>>>(
        pf2, voxfeat1, idx, Wt3, b3, pf4);
    k_point_mlp4<<<NPTS / 128, 256, 256 * 256 * 2, stream>>>(pf4, idx, Wt4, b4,
                                                             (unsigned int*)vox2buf);
    k_vox_mlp2<<<NVOX / 128, 256, 256 * 256 * 2, stream>>>(vox2buf, Wtv2, bv2, out);
}
